// TBNN_Q_direction_88081189307077
// MI455X (gfx1250) — compile-verified, matrix operand order checked by reading, not run on hardware
//
#include <hip/hip_runtime.h>

// TBNN Q-direction kernel for gfx1250 (MI455X), wave32 + WMMA f16->f32.
//
// Roofline: ~32 GFLOP (MLP GEMMs) vs ~57 MB HBM traffic -> compute bound.
// One wave = 16-sample M-tile; MLP layers as v_wmma_f32_16x16x32_f16 against
// block-shared f16 weights (transposed + zero-padded in LDS).
// v3: fragments loaded as two aligned b128s (v8h) instead of 4x ds_load_2addr,
// weight rows padded to KP+8 f16 so B-fragment loads are bank-conflict-free
// (stride 272B -> lane n hits bank 4n for the 128-wide layers).

typedef __attribute__((ext_vector_type(16))) _Float16 v16h;
typedef __attribute__((ext_vector_type(8)))  _Float16 v8h;
typedef __attribute__((ext_vector_type(8)))  float    v8f;

#define HSTRIDE 136  // f16 elements per activation row (128 + 8 pad)

// ---- fragment loader: 16-bit A/B matrix layout (ISA 7.12.2) -----------------
// Lane's 16 elements = 8 consecutive f16 at kBase+half*8 plus 8 consecutive
// f16 at kBase+16+half*8 -> two aligned 16-byte LDS loads.
__device__ inline v16h ld_frag(const _Float16* __restrict__ row, int kBase, int half) {
  v8h lo = *(const v8h*)(row + kBase + half * 8);
  v8h hi = *(const v8h*)(row + kBase + 16 + half * 8);
  v16h f;
#pragma unroll
  for (int i = 0; i < 8; ++i) {
    f[i]     = lo[i];
    f[8 + i] = hi[i];
  }
  return f;
}

// ---- one MLP layer: [16 x KP] (f16, LDS) x [KP x NP] (f16, LDS) + bias ------
// Weights stored transposed [NP][KP+8] f16 (row pad kills bank conflicts).
template <int KP, int NP, bool LAST>
__device__ inline void mlp_layer(const _Float16* __restrict__ wT,
                                 const float* __restrict__ bias,
                                 const _Float16* __restrict__ hIn,
                                 _Float16* __restrict__ hOut,
                                 float* __restrict__ gOut,
                                 int lane) {
  const int nl   = lane & 15;
  const int half = lane >> 4;
  const int NT   = NP / 16;
  const int KT   = KP / 32;
  const int WS   = KP + 8;  // padded weight row stride

  // A fragments are invariant across N-tiles: load once into registers.
  v16h a[KT];
  {
    const _Float16* aRow = hIn + nl * HSTRIDE;
#pragma unroll
    for (int kt = 0; kt < KT; ++kt) a[kt] = ld_frag(aRow, kt * 32, half);
  }

  // Process N-tiles in pairs -> two independent WMMA accumulation chains.
#pragma unroll
  for (int nt = 0; nt < NT; nt += 2) {
    const bool two = (nt + 1 < NT);
    const _Float16* bRow0 = wT + (nt * 16 + nl) * WS;
    const _Float16* bRow1 = bRow0 + 16 * WS;
    float bv0 = bias[nt * 16 + nl];
    v8f c0 = {bv0, bv0, bv0, bv0, bv0, bv0, bv0, bv0};
    v8f c1 = {};
    if (two) {
      float bv1 = bias[(nt + 1) * 16 + nl];
      c1 = (v8f){bv1, bv1, bv1, bv1, bv1, bv1, bv1, bv1};
    }
#pragma unroll
    for (int kt = 0; kt < KT; ++kt) {
      v16h b0 = ld_frag(bRow0, kt * 32, half);
      c0 = __builtin_amdgcn_wmma_f32_16x16x32_f16(false, a[kt], false, b0, (short)0,
                                                  c0, false, false);
      if (two) {
        v16h b1 = ld_frag(bRow1, kt * 32, half);
        c1 = __builtin_amdgcn_wmma_f32_16x16x32_f16(false, a[kt], false, b1, (short)0,
                                                    c1, false, false);
      }
    }
#pragma unroll
    for (int r = 0; r < 8; ++r) {
      int m = r + (half << 3);  // D layout: VGPR r -> M=r / r+8
      if (LAST) {
        gOut[m * 16 + nl] = c0[r];
      } else {
        float x0 = fmaxf(c0[r], 0.1f * c0[r]);  // leaky relu, slope in (0,1)
        hOut[m * HSTRIDE + nt * 16 + nl] = (_Float16)x0;
        if (two) {
          float x1 = fmaxf(c1[r], 0.1f * c1[r]);
          hOut[m * HSTRIDE + (nt + 1) * 16 + nl] = (_Float16)x1;
        }
      }
    }
  }
}

// ---- 3x3 helpers ------------------------------------------------------------
__device__ inline void mm3(const float* A, const float* B, float* C) {
#pragma unroll
  for (int i = 0; i < 3; ++i)
#pragma unroll
    for (int j = 0; j < 3; ++j)
      C[3 * i + j] = A[3 * i] * B[j] + A[3 * i + 1] * B[3 + j] + A[3 * i + 2] * B[6 + j];
}
__device__ inline float trprod(const float* A, const float* B) {  // tr(A*B)
  float t = 0.f;
#pragma unroll
  for (int i = 0; i < 3; ++i)
#pragma unroll
    for (int j = 0; j < 3; ++j) t += A[3 * i + j] * B[3 * j + i];
  return t;
}

__global__ __launch_bounds__(256) void tbnn_kernel(
    const float* __restrict__ sp, const float* __restrict__ wp,
    const float* __restrict__ W0, const float* __restrict__ b0,
    const float* __restrict__ W1, const float* __restrict__ b1,
    const float* __restrict__ W2, const float* __restrict__ b2,
    const float* __restrict__ W3, const float* __restrict__ b3,
    const float* __restrict__ W4, const float* __restrict__ b4,
    const float* __restrict__ W5, const float* __restrict__ b5,
    float* __restrict__ outp, int n) {
  extern __shared__ char smem[];
  // LDS map: [0,101888)       weights f16 transposed, rows padded to KP+8
  //          [101888,103808)  biases f32 padded
  //          [103808,173440)  per-wave h ping-pong (8 waves x 2 x 16x136 f16)
  //          [173440,181632)  per-wave g staging (8 waves x 16x16 f32)
  _Float16* wbuf = (_Float16*)smem;
  float*    bbuf = (float*)(smem + 101888);
  const int waveId = threadIdx.x >> 5;
  const int lane   = threadIdx.x & 31;
  _Float16* h0   = (_Float16*)(smem + 103808 + waveId * 8704);
  _Float16* h1   = h0 + 2176;
  float*    gbuf = (float*)(smem + 173440 + waveId * 1024);

  const float* Wg[6]  = {W0, W1, W2, W3, W4, W5};
  const float* Bg[6]  = {b0, b1, b2, b3, b4, b5};
  const int KD[6]   = {5, 50, 100, 100, 100, 50};
  const int ND[6]   = {50, 100, 100, 100, 50, 10};
  const int WSA[6]  = {40, 72, 136, 136, 136, 72};            // padded row strides
  const int WOFF[6] = {0, 2560, 10624, 25856, 41088, 49792};  // f16 offsets
  const int BOFF[6] = {0, 64, 176, 288, 400, 464};            // f32 offsets

  // ---- stage weights: zero padded region, then scatter transposed f16 ----
  for (int i = threadIdx.x; i < 25472; i += blockDim.x) ((unsigned*)wbuf)[i] = 0u;
  for (int i = threadIdx.x; i < 480; i += blockDim.x) bbuf[i] = 0.f;
  __syncthreads();
  for (int l = 0; l < 6; ++l) {
    const int K = KD[l], N = ND[l], WS = WSA[l];
    const float* Wsrc = Wg[l];
    for (int i = threadIdx.x; i < K * N; i += blockDim.x) {
      int k = i / N, nn = i - k * N;
      wbuf[WOFF[l] + nn * WS + k] = (_Float16)Wsrc[i];
    }
    const float* bsrc = Bg[l];
    for (int i = threadIdx.x; i < N; i += blockDim.x) bbuf[BOFF[l] + i] = bsrc[i];
  }
  // zero both activation buffers once; padded K columns (>=112) stay 0 forever
  {
    unsigned* hz = (unsigned*)h0;
    for (int i = lane; i < 2176; i += 32) hz[i] = 0u;
  }
  __syncthreads();

  const int totalWaves  = gridDim.x * (blockDim.x >> 5);
  const int waveGlobal  = blockIdx.x * (blockDim.x >> 5) + waveId;
  const int totalGroups = (n + 15) >> 4;
  const int iters = (totalGroups + totalWaves - 1) / totalWaves;

  for (int it = 0; it < iters; ++it) {
    const int base = (waveGlobal + it * totalWaves) << 4;
    const bool active = base < n;

    // clear invariant staging region of h0 (rows 0-15, f16 cols 0-31)
    if (active) {
      unsigned* hz = (unsigned*)h0;
      for (int i = lane; i < 256; i += 32) hz[(i >> 4) * 68 + (i & 15)] = 0u;
    }
    __syncthreads();

    float sm[9], wm[9], s2[9], w2[9], l1 = 0.f, l2 = 0.f;
    const int si = base + (lane & 15);
    const bool doS = active && (lane < 16) && (si < n);
    if (doS) {
      __builtin_prefetch(sp + (size_t)(si + 16 * totalWaves) * 9, 0, 1);
      __builtin_prefetch(wp + (size_t)(si + 16 * totalWaves) * 9, 0, 1);
#pragma unroll
      for (int i = 0; i < 9; ++i) {
        sm[i] = sp[si * 9 + i];
        wm[i] = wp[si * 9 + i];
      }
      mm3(sm, sm, s2);
      mm3(wm, wm, w2);
      l1 = s2[0] + s2[4] + s2[8];
      l2 = w2[0] + w2[4] + w2[8];
      float l3 = trprod(s2, sm);
      float l4 = trprod(w2, sm);
      float l5 = trprod(w2, s2);
      _Float16* row = h0 + (lane & 15) * HSTRIDE;
      row[0] = (_Float16)l1;
      row[1] = (_Float16)l2;
      row[2] = (_Float16)l3;
      row[3] = (_Float16)l4;
      row[4] = (_Float16)l5;
    }
    __syncthreads();

    // ---- MLP: 5->50->100->100->100->50->10, WMMA f16 in / f32 acc ----
    if (active) mlp_layer<32, 64, false>(wbuf + WOFF[0], bbuf + BOFF[0], h0, h1, nullptr, lane);
    __syncthreads();
    if (active) mlp_layer<64, 112, false>(wbuf + WOFF[1], bbuf + BOFF[1], h1, h0, nullptr, lane);
    __syncthreads();
    if (active) mlp_layer<128, 112, false>(wbuf + WOFF[2], bbuf + BOFF[2], h0, h1, nullptr, lane);
    __syncthreads();
    if (active) mlp_layer<128, 112, false>(wbuf + WOFF[3], bbuf + BOFF[3], h1, h0, nullptr, lane);
    __syncthreads();
    if (active) mlp_layer<128, 64, false>(wbuf + WOFF[4], bbuf + BOFF[4], h0, h1, nullptr, lane);
    __syncthreads();
    if (active) mlp_layer<64, 16, true>(wbuf + WOFF[5], bbuf + BOFF[5], h1, nullptr, gbuf, lane);
    __syncthreads();

    // ---- combine: Q = sum g_n T_n, deviatoric, normalize ----
    if (doS) {
      float g[10];
#pragma unroll
      for (int i = 0; i < 10; ++i) g[i] = gbuf[(lane & 15) * 16 + i];
      float sw[9], ws[9], ws2[9], s2w[9], w2s[9], sw2[9], s2w2[9], w2s2[9];
      float t7a[9], t7b[9], t8a[9], t8b[9], t10a[9], t10b[9];
      mm3(sm, wm, sw);
      mm3(wm, sm, ws);
      mm3(wm, s2, ws2);
      mm3(s2, wm, s2w);
      mm3(w2, sm, w2s);
      mm3(sm, w2, sw2);
      mm3(s2, w2, s2w2);
      mm3(w2, s2, w2s2);
      mm3(ws, w2, t7a);
      mm3(w2, sw, t7b);
      mm3(sw, s2, t8a);
      mm3(s2, ws, t8b);
      mm3(ws2, w2, t10a);
      mm3(w2, s2w, t10b);
      float trsw2  = sw2[0] + sw2[4] + sw2[8];
      float trs2w2 = s2w2[0] + s2w2[4] + s2w2[8];
      float q[9];
#pragma unroll
      for (int i = 0; i < 9; ++i) {
        float d = (i == 0 || i == 4 || i == 8) ? 1.f : 0.f;
        q[i] = g[0] * sm[i]
             + g[1] * (sw[i] - ws[i])
             + g[2] * (s2[i] - (l1 * (1.f / 3.f)) * d)
             + g[3] * (w2[i] - (l2 * (1.f / 3.f)) * d)
             + g[4] * (ws2[i] - s2w[i])
             + g[5] * (w2s[i] + sw2[i] - (2.f / 3.f) * trsw2 * d)
             + g[6] * (t7a[i] - t7b[i])
             + g[7] * (t8a[i] - t8b[i])
             + g[8] * (w2s2[i] + s2w2[i] - (2.f / 3.f) * trs2w2 * d)
             + g[9] * (t10a[i] - t10b[i]);
      }
      float trq = (q[0] + q[4] + q[8]) * (1.f / 3.f);
      q[0] -= trq; q[4] -= trq; q[8] -= trq;
      float ss = 0.f;
#pragma unroll
      for (int i = 0; i < 9; ++i) ss += q[i] * q[i];
      float nrm = sqrtf(ss);
      nrm = (nrm == 0.f) ? 1e-8f : nrm;
      float inv = 1.f / nrm;
#pragma unroll
      for (int i = 0; i < 9; ++i) outp[si * 9 + i] = q[i] * inv;
    }
    __syncthreads();
  }
}

extern "C" void kernel_launch(void* const* d_in, const int* in_sizes, int n_in,
                              void* d_out, int out_size, void* d_ws, size_t ws_size,
                              hipStream_t stream) {
  const float* s  = (const float*)d_in[0];
  const float* w  = (const float*)d_in[1];
  const float* W0 = (const float*)d_in[2];
  const float* b0 = (const float*)d_in[3];
  const float* W1 = (const float*)d_in[4];
  const float* b1 = (const float*)d_in[5];
  const float* W2 = (const float*)d_in[6];
  const float* b2 = (const float*)d_in[7];
  const float* W3 = (const float*)d_in[8];
  const float* b3 = (const float*)d_in[9];
  const float* W4 = (const float*)d_in[10];
  const float* b4 = (const float*)d_in[11];
  const float* W5 = (const float*)d_in[12];
  const float* b5 = (const float*)d_in[13];
  float* out = (float*)d_out;
  const int n = in_sizes[0] / 9;

  const int block = 256;                 // 8 wave32 per block (one WGP)
  const int wavesPerBlock = block / 32;
  int groups = (n + 15) / 16;
  int grid = (groups + wavesPerBlock - 1) / wavesPerBlock;
  if (grid > 1024) grid = 1024;          // grid-stride, amortizes weight staging
  if (grid < 1) grid = 1;
  const size_t lds = 181632;             // ~177 KB dynamic LDS (WGP has 320 KB)

  hipFuncSetAttribute(reinterpret_cast<const void*>(tbnn_kernel),
                      hipFuncAttributeMaxDynamicSharedMemorySize, (int)lds);
  hipLaunchKernelGGL(tbnn_kernel, dim3(grid), dim3(block), lds, stream,
                     s, w, W0, b0, W1, b1, W2, b2, W3, b3, W4, b4, W5, b5, out, n);
}
